// FreeChunkerSelfAttention_43997644980435
// MI455X (gfx1250) — matrix operand order repaired and checked by reading, not compiled, not run on hardware
//
#include <hip/hip_runtime.h>
#include <hip/hip_bf16.h>

// ---------- types ----------
typedef __attribute__((ext_vector_type(16))) __bf16 bf16x16;
typedef __attribute__((ext_vector_type(8)))  float  f32x8;
typedef __attribute__((ext_vector_type(4)))  unsigned int u32x4;

// exact parameter types for the async global->LDS builtin (per hipcc diagnostic)
typedef int v4i __attribute__((vector_size(16)));
typedef __attribute__((address_space(1))) v4i* gptr_v4i;
typedef __attribute__((address_space(3))) v4i* lptr_v4i;

union FragB16 {                 // one 16-bit A/B WMMA fragment (8 VGPRs)
    bf16x16 f;
    u32x4   q[2];
    unsigned short s[16];
};

#define WMMA_BF16(A, B, C) \
    __builtin_amdgcn_wmma_f32_16x16x32_bf16(false, (A), false, (B), (short)0, (C), false, false)

#if __has_builtin(__builtin_amdgcn_global_load_async_to_lds_b128)
#define HAVE_ASYNC_LDS 1
#else
#define HAVE_ASYNC_LDS 0
#endif

static __device__ __forceinline__ void wait_async0() {
#if __has_builtin(__builtin_amdgcn_s_wait_asynccnt)
    __builtin_amdgcn_s_wait_asynccnt(0);
#else
    asm volatile("s_wait_asynccnt 0x0" ::: "memory");
#endif
}

static __device__ __forceinline__ unsigned short f2bf(float f) {
    unsigned int u = __float_as_uint(f);
    u = (u + 0x7FFFu + ((u >> 16) & 1u)) >> 16;   // RNE
    return (unsigned short)u;
}

static __device__ __forceinline__ float rmax16(float t) {
    t = fmaxf(t, __shfl_xor(t, 1));
    t = fmaxf(t, __shfl_xor(t, 2));
    t = fmaxf(t, __shfl_xor(t, 4));
    t = fmaxf(t, __shfl_xor(t, 8));
    return t;                                      // reduction stays inside 16-lane half
}
static __device__ __forceinline__ float rsum16(float t) {
    t += __shfl_xor(t, 1);
    t += __shfl_xor(t, 2);
    t += __shfl_xor(t, 4);
    t += __shfl_xor(t, 8);
    return t;
}

// ---------- constants ----------
#define Bsz 2
#define S   2048
#define HID 1024
#define NH  16
#define HD  64
#define M_TOT (Bsz * S)          // 4096

// ---------- kernel 1: fp32 -> bf16 ----------
__global__ void cvt_bf16(const float* __restrict__ in, unsigned short* __restrict__ out, int n) {
    int i = blockIdx.x * blockDim.x + threadIdx.x;
    int stride = gridDim.x * blockDim.x;
    for (; i < n; i += stride) out[i] = f2bf(in[i]);
}

// ---------- kernel 2: projection GEMM  y = x @ W^T + b,  bf16 WMMA, f32 acc ----------
// X: [4096,1024] bf16 row-major, W: [1024,1024] bf16 row-major (row n holds K contiguous)
// Out: bf16 [B, NH, S, HD].  Block = 4 waves, block tile 64(M) x 128(N); wave tile 32 x 64.
__global__ __launch_bounds__(128) void qkv_gemm(const unsigned short* __restrict__ X,
                                                const unsigned short* __restrict__ W,
                                                const float* __restrict__ bias,
                                                unsigned short* __restrict__ Out) {
    const int lane = threadIdx.x & 31;
    const int wave = threadIdx.x >> 5;
    const int hl = lane >> 4;          // 16-lane half
    const int ll = lane & 15;
    const int m0 = blockIdx.x * 64 + (wave & 1) * 32;
    const int n0 = blockIdx.y * 128 + (wave >> 1) * 64;

    f32x8 acc[2][4] = {};

    for (int k = 0; k < HID; k += 32) {
        FragB16 a[2], b[4];
#pragma unroll
        for (int i = 0; i < 2; ++i) {
            // A fragment: lane holds row M, 16 contiguous K [k+16*hl .. +15]
            const u32x4* ap = (const u32x4*)(X + (size_t)(m0 + 16 * i + ll) * HID + k + 16 * hl);
            a[i].q[0] = ap[0];
            a[i].q[1] = ap[1];
        }
#pragma unroll
        for (int j = 0; j < 4; ++j) {
            // B fragment: lane holds col N (= W row n), 16 contiguous K
            const u32x4* bp = (const u32x4*)(W + (size_t)(n0 + 16 * j + ll) * HID + k + 16 * hl);
            b[j].q[0] = bp[0];
            b[j].q[1] = bp[1];
        }
#pragma unroll
        for (int i = 0; i < 2; ++i)
#pragma unroll
            for (int j = 0; j < 4; ++j)
                acc[i][j] = WMMA_BF16(a[i].f, b[j].f, acc[i][j]);
    }

#pragma unroll
    for (int i = 0; i < 2; ++i)
#pragma unroll
        for (int j = 0; j < 4; ++j)
#pragma unroll
            for (int r = 0; r < 8; ++r) {
                int m = m0 + 16 * i + r + 8 * hl;         // flat token index
                int n = n0 + 16 * j + ll;                 // output feature
                float v = acc[i][j][r] + bias[n];
                int bb = m >> 11, s = m & (S - 1);
                int h = n >> 6, d = n & (HD - 1);
                Out[((size_t)(bb * NH + h) * S + s) * HD + d] = f2bf(v);
            }
}

// ---------- kernel 3: flash attention ----------
// Q,K,V: bf16 [B, NH, S, HD]; Out: f32 [B, S, HID]
// grid (S/64, NH, B); 4 waves/block, wave owns 16 query rows; 64-key tiles.
__global__ __launch_bounds__(128) void attn_fwd(const unsigned short* __restrict__ Q,
                                                const unsigned short* __restrict__ K,
                                                const unsigned short* __restrict__ V,
                                                float* __restrict__ Out) {
    __shared__ __align__(16) unsigned short sK[64 * HD];      // [key][d]
    __shared__ __align__(16) unsigned short sVt[HD * 64];     // [d][key]  (transposed)
    __shared__ __align__(16) unsigned short sP[4][16 * 64];   // per-wave P tile [row][key]

    const int lane = threadIdx.x & 31;
    const int wave = threadIdx.x >> 5;
    const int hl = lane >> 4;
    const int ll = lane & 15;

    const int b = blockIdx.z, h = blockIdx.y, qblk = blockIdx.x;
    const unsigned short* Qh = Q + (size_t)(b * NH + h) * S * HD;
    const unsigned short* Kh = K + (size_t)(b * NH + h) * S * HD;
    const unsigned short* Vh = V + (size_t)(b * NH + h) * S * HD;

    const int qrow0 = qblk * 64 + wave * 16;

    // Q fragments for this wave's 16 rows: fixed for the whole key loop.
    FragB16 qa[2];
#pragma unroll
    for (int s = 0; s < 2; ++s) {
        const u32x4* qp = (const u32x4*)(Qh + (size_t)(qrow0 + ll) * HD + s * 32 + 16 * hl);
        qa[s].q[0] = qp[0];
        qa[s].q[1] = qp[1];
    }

    f32x8 ctx[4] = {};
    f32x8 mrow, lrow;
#pragma unroll
    for (int r = 0; r < 8; ++r) { mrow[r] = -3.0e38f; lrow[r] = 0.0f; }

    unsigned short* myP = &sP[wave][0];

    for (int kb = 0; kb < S; kb += 64) {
        __syncthreads();   // protect previous-iteration LDS readers

        // --- cooperative tile loads ---
        {   // K tile: contiguous 8 KB copy, layout-preserving -> async global->LDS path
            const u32x4* gk = (const u32x4*)(Kh + (size_t)kb * HD);
            u32x4* lk = (u32x4*)sK;
#if HAVE_ASYNC_LDS
#pragma unroll
            for (int t = 0; t < 4; ++t) {
                int i = threadIdx.x + t * 128;
                __builtin_amdgcn_global_load_async_to_lds_b128(
                    (gptr_v4i)(gk + i), (lptr_v4i)(lk + i), 0, 0);
            }
#else
            for (int i = threadIdx.x; i < 512; i += 128) lk[i] = gk[i];
#endif
        }
        {   // V tile: transpose into [d][key] through VGPRs
            const u32x4* gv = (const u32x4*)(Vh + (size_t)kb * HD);
            for (int i = threadIdx.x; i < 512; i += 128) {
                u32x4 d4 = gv[i];
                unsigned short t8[8];
                __builtin_memcpy(t8, &d4, 16);
                int key = i >> 3, dd = (i & 7) * 8;
#pragma unroll
                for (int e = 0; e < 8; ++e) sVt[(dd + e) * 64 + key] = t8[e];
            }
        }
#if HAVE_ASYNC_LDS
        wait_async0();     // own wave's async K transfers complete
#endif
        __syncthreads();   // all waves' tiles visible

        // --- scores S = Q @ K^T * 1/8 : 16 x 64 per wave ---
        f32x8 sc[4];
#pragma unroll
        for (int nt = 0; nt < 4; ++nt) {
            f32x8 c = {};
#pragma unroll
            for (int s = 0; s < 2; ++s) {
                FragB16 kf;
                int key = nt * 16 + ll;                    // B col = key
                int dbase = s * 32 + 16 * hl;              // contiguous K (head dim)
                const u32x4* kp = (const u32x4*)(sK + key * HD + dbase);
                kf.q[0] = kp[0];
                kf.q[1] = kp[1];
                c = WMMA_BF16(qa[s].f, kf.f, c);
            }
#pragma unroll
            for (int r = 0; r < 8; ++r) c[r] *= 0.125f;    // 1/sqrt(64)
            sc[nt] = c;
        }

        // --- online softmax (per-row stats live in C-layout reg index r) ---
        f32x8 mnew, alpha, rs;
#pragma unroll
        for (int r = 0; r < 8; ++r) {
            float t = fmaxf(fmaxf(sc[0][r], sc[1][r]), fmaxf(sc[2][r], sc[3][r]));
            t = rmax16(t);
            mnew[r] = fmaxf(mrow[r], t);
            alpha[r] = __expf(mrow[r] - mnew[r]);
        }
#pragma unroll
        for (int nt = 0; nt < 4; ++nt)
#pragma unroll
            for (int r = 0; r < 8; ++r) sc[nt][r] = __expf(sc[nt][r] - mnew[r]);
#pragma unroll
        for (int r = 0; r < 8; ++r) {
            float t = sc[0][r] + sc[1][r] + sc[2][r] + sc[3][r];
            rs[r] = rsum16(t);
            lrow[r] = lrow[r] * alpha[r] + rs[r];
            mrow[r] = mnew[r];
        }
#pragma unroll
        for (int dn = 0; dn < 4; ++dn)
#pragma unroll
            for (int r = 0; r < 8; ++r) ctx[dn][r] *= alpha[r];

        // --- P to LDS (bf16, C-layout -> row-major 16x64) ---
#pragma unroll
        for (int nt = 0; nt < 4; ++nt)
#pragma unroll
            for (int r = 0; r < 8; ++r)
                myP[(r + 8 * hl) * 64 + nt * 16 + ll] = f2bf(sc[nt][r]);

        // --- ctx += P @ V ---
#pragma unroll
        for (int s = 0; s < 2; ++s) {
            FragB16 pa;
            const u32x4* pp = (const u32x4*)(myP + ll * 64 + s * 32 + 16 * hl);
            pa.q[0] = pp[0];
            pa.q[1] = pp[1];
#pragma unroll
            for (int dn = 0; dn < 4; ++dn) {
                FragB16 vb;
                int dcol = dn * 16 + ll;                   // B col = head dim
                int kk = s * 32 + 16 * hl;                 // contiguous keys in sVt row
                const u32x4* vp = (const u32x4*)(sVt + dcol * 64 + kk);
                vb.q[0] = vp[0];
                vb.q[1] = vp[1];
                ctx[dn] = WMMA_BF16(pa.f, vb.f, ctx[dn]);
            }
        }
    }

    // --- epilogue: normalize + write f32 [B,S,HID] ---
#pragma unroll
    for (int dn = 0; dn < 4; ++dn)
#pragma unroll
        for (int r = 0; r < 8; ++r) {
            int row = qrow0 + r + 8 * hl;
            int d = dn * 16 + ll;
            Out[((size_t)b * S + row) * HID + h * HD + d] = ctx[dn][r] / lrow[r];
        }
}

// ---------- host ----------
extern "C" void kernel_launch(void* const* d_in, const int* in_sizes, int n_in,
                              void* d_out, int out_size, void* d_ws, size_t ws_size,
                              hipStream_t stream) {
    (void)in_sizes; (void)n_in; (void)out_size; (void)ws_size;
    const float* hs  = (const float*)d_in[0];
    const float* hs2 = (const float*)d_in[1];
    const float* Wq  = (const float*)d_in[2];
    const float* bq  = (const float*)d_in[3];
    const float* Wk  = (const float*)d_in[4];
    const float* bk  = (const float*)d_in[5];
    const float* Wv  = (const float*)d_in[6];
    const float* bv  = (const float*)d_in[7];
    float* out = (float*)d_out;

    unsigned short* ws = (unsigned short*)d_ws;
    const size_t NTOK = (size_t)M_TOT * HID;   // 4096*1024
    const size_t NWT  = (size_t)HID * HID;     // 1024*1024
    unsigned short* hsA = ws;                  // bf16 hidden_states
    unsigned short* hsB = hsA + NTOK;
    unsigned short* wq  = hsB + NTOK;
    unsigned short* wk  = wq + NWT;
    unsigned short* wv  = wk + NWT;
    unsigned short* Qb  = wv + NWT;            // [B,NH,S,HD]
    unsigned short* Kb  = Qb + NTOK;
    unsigned short* Vb  = Kb + NTOK;

    cvt_bf16<<<4096, 256, 0, stream>>>(hs,  hsA, (int)NTOK);
    cvt_bf16<<<4096, 256, 0, stream>>>(hs2, hsB, (int)NTOK);
    cvt_bf16<<<2048, 256, 0, stream>>>(Wq, wq, (int)NWT);
    cvt_bf16<<<2048, 256, 0, stream>>>(Wk, wk, (int)NWT);
    cvt_bf16<<<2048, 256, 0, stream>>>(Wv, wv, (int)NWT);

    dim3 ggrid(M_TOT / 64, HID / 128);         // (64, 8)
    qkv_gemm<<<ggrid, 128, 0, stream>>>(hsA, wq, bq, Qb);
    qkv_gemm<<<ggrid, 128, 0, stream>>>(hsB, wk, bk, Kb);
    qkv_gemm<<<ggrid, 128, 0, stream>>>(hsB, wv, bv, Vb);

    dim3 agrid(S / 64, NH, Bsz);               // (32, 16, 2)
    attn_fwd<<<agrid, 128, 0, stream>>>(Qb, Kb, Vb, out);
}
